// SRLGCN_56418690400424
// MI455X (gfx1250) — matrix operand order repaired and checked by reading, not compiled, not run on hardware
//
#include <hip/hip_runtime.h>
#include <stdint.h>

// ---------------------------------------------------------------------------
// GCN forward on MI455X (gfx1250, wave32).
// GEMMs:   bf16 V_WMMA_F32_16X16X32_BF16, f32 accumulate.
// Message passing: f32 global_atomic_add_f32 (h + agg buffers are L2-resident:
//                  51.2MB each << 192MB L2).
// ---------------------------------------------------------------------------

#define N_NODES   50000
#define N_EDGES   800000
#define IN_DIM    768
#define HIDDEN    256
#define N_CLASSES 10
#define N_GRAPHS  128

typedef __attribute__((ext_vector_type(16))) __bf16          v16bf;
typedef __attribute__((ext_vector_type(16))) unsigned short  v16u;
typedef __attribute__((ext_vector_type(8)))  unsigned short  v8u;
typedef __attribute__((ext_vector_type(8)))  float           v8f;

__device__ __forceinline__ unsigned short f2bf(float f) {
    // round-to-nearest-even f32 -> bf16
    unsigned int u = __builtin_bit_cast(unsigned int, f);
    u += 0x7fffu + ((u >> 16) & 1u);
    return (unsigned short)(u >> 16);
}

// ---------------------------------------------------------------------------
// Elementwise helpers
// ---------------------------------------------------------------------------
__global__ void zero_f32(float* p, long long n) {
    long long i = (long long)blockIdx.x * blockDim.x + threadIdx.x;
    long long stride = (long long)gridDim.x * blockDim.x;
    for (; i < n; i += stride) p[i] = 0.0f;
}

__global__ void cvt_f32_bf16(const float* __restrict__ src, unsigned short* __restrict__ dst,
                             long long n) {
    long long i = (long long)blockIdx.x * blockDim.x + threadIdx.x;
    long long stride = (long long)gridDim.x * blockDim.x;
    for (; i < n; i += stride) dst[i] = f2bf(src[i]);
}

// W[K][N] (f32, row-major) -> Wt[N][K] (bf16): each WMMA lane then reads its
// output column's K-run contiguously.
__global__ void w_transpose_bf16(const float* __restrict__ W, unsigned short* __restrict__ Wt,
                                 int K, int N) {
    long long i = (long long)blockIdx.x * blockDim.x + threadIdx.x;
    if (i >= (long long)K * N) return;
    int k = (int)(i / N), n = (int)(i % N);
    Wt[(long long)n * K + k] = f2bf(W[i]);
}

// ---------------------------------------------------------------------------
// Degree / normalization  (self-loop folded into init: deg starts at 1)
// ---------------------------------------------------------------------------
__global__ void deg_init(float* deg, int n) {
    int i = blockIdx.x * blockDim.x + threadIdx.x;
    if (i < n) deg[i] = 1.0f;
}
__global__ void deg_edges(const int* __restrict__ dst, float* deg, int nE) {
    int e = blockIdx.x * blockDim.x + threadIdx.x;
    if (e < nE) atomicAdd(&deg[dst[e]], 1.0f);
}
__global__ void dinv_kernel(const float* __restrict__ deg, float* dinv, int n) {
    int i = blockIdx.x * blockDim.x + threadIdx.x;
    if (i < n) dinv[i] = rsqrtf(deg[i]);   // deg >= 1 always (self-loop)
}

// ---------------------------------------------------------------------------
// bf16 WMMA GEMM:  C[M][N] = A[M][K] * Bt[N][K]^T     (M%16==0, N%16==0, K%32==0)
// One wave -> one 16x16 f32 tile. 256 threads = 8 waves/block.
// ---------------------------------------------------------------------------
__global__ __launch_bounds__(256) void gemm_bf16_wmma(
    const unsigned short* __restrict__ A,   // [M x K] bf16 row-major
    const unsigned short* __restrict__ Bt,  // [N x K] bf16 (col n contiguous in K)
    float* __restrict__ C,                  // [M x N] f32 row-major
    int M, int N, int K)
{
    const int lane = threadIdx.x & 31;
    const int wave = threadIdx.x >> 5;
    const int tilesN = N >> 4;
    const int tile = blockIdx.x * 8 + wave;
    const int tm = (tile / tilesN) << 4;    // uniform per wave -> EXEC stays all-1s
    const int tn = (tile % tilesN) << 4;
    if (tm >= M) return;

    const int half = lane >> 4;             // 0: lanes 0-15, 1: lanes 16-31
    const int l15  = lane & 15;

    // A: lane (m = tm+l15) reads K = k0+16h .. k0+16h+15 contiguously
    const unsigned short* arow = A  + (long long)(tm + l15) * K + half * 16;
    // B: lane (n = tn+l15) reads K = k0+16h .. k0+16h+15 contiguously
    const unsigned short* bcol = Bt + (long long)(tn + l15) * K + half * 16;

    v8f acc = {};
    for (int k0 = 0; k0 < K; k0 += 32) {
        __builtin_prefetch((const void*)(arow + k0 + 128), 0, 0);  // global_prefetch_b8
        v8u a0 = *(const v8u*)(arow + k0);
        v8u a1 = *(const v8u*)(arow + k0 + 8);
        v16u au, bu;
        bu = *(const v16u*)(bcol + k0);
#pragma unroll
        for (int i = 0; i < 8; ++i) { au[i] = a0[i]; au[i + 8] = a1[i]; }
        v16bf av = __builtin_bit_cast(v16bf, au);
        v16bf bv = __builtin_bit_cast(v16bf, bu);
        acc = __builtin_amdgcn_wmma_f32_16x16x32_bf16(
                  false, av, false, bv, (short)0, acc, false, false);
    }

    float* crow = C + (long long)(tm + half * 8) * N + tn + l15;
#pragma unroll
    for (int r = 0; r < 8; ++r) crow[(long long)r * N] = acc[r];
}

// ---------------------------------------------------------------------------
// Message passing: out[dst] += h[src] * dinv[src]*dinv[dst]   (one wave/edge)
// ---------------------------------------------------------------------------
__global__ __launch_bounds__(256) void agg_edges(
    const float* __restrict__ h, const int* __restrict__ src, const int* __restrict__ dst,
    const float* __restrict__ dinv, float* out, int nE)
{
    int e = blockIdx.x * 8 + (threadIdx.x >> 5);
    int lane = threadIdx.x & 31;
    if (e >= nE) return;
    int s = src[e], d = dst[e];
    float norm = dinv[s] * dinv[d];
    const float* hs = h + (long long)s * HIDDEN;
    float* od = out + (long long)d * HIDDEN;
#pragma unroll
    for (int i = 0; i < HIDDEN / 32; ++i) {
        int f = lane + i * 32;
        atomicAdd(&od[f], hs[f] * norm);
    }
}

// self-loop contribution (runs after agg_edges on the stream -> plain RMW)
__global__ void self_loops(const float* __restrict__ h, const float* __restrict__ dinv,
                           float* out) {
    int n = blockIdx.x, f = threadIdx.x;
    float w = dinv[n] * dinv[n];
    long long i = (long long)n * HIDDEN + f;
    out[i] += h[i] * w;
}

__global__ void bias_act(float* x, const float* __restrict__ b, int relu) {
    int n = blockIdx.x, f = threadIdx.x;
    long long i = (long long)n * HIDDEN + f;
    float v = x[i] + b[f];
    x[i] = relu ? fmaxf(v, 0.0f) : v;
}

// ---------------------------------------------------------------------------
// Pooling + FC head
// ---------------------------------------------------------------------------
__global__ void pool_nodes(const float* __restrict__ x, const int* __restrict__ batch,
                           float* pooled, float* cnt) {
    int n = blockIdx.x, f = threadIdx.x;
    int g = batch[n];
    atomicAdd(&pooled[(long long)g * HIDDEN + f], x[(long long)n * HIDDEN + f]);
    if (f == 0) atomicAdd(&cnt[g], 1.0f);
}

__global__ void final_fc(const float* __restrict__ pooled, const float* __restrict__ cnt,
                         const float* __restrict__ Wfc, const float* __restrict__ bfc,
                         float* out) {
    int g = blockIdx.x, c = threadIdx.x;
    if (c >= N_CLASSES) return;
    float inv = 1.0f / fmaxf(cnt[g], 1.0f);
    float s = bfc[c];
    for (int f = 0; f < HIDDEN; ++f)
        s += pooled[(long long)g * HIDDEN + f] * inv * Wfc[f * N_CLASSES + c];
    out[g * N_CLASSES + c] = s;
}

// ---------------------------------------------------------------------------
// Host launcher
// ---------------------------------------------------------------------------
extern "C" void kernel_launch(void* const* d_in, const int* in_sizes, int n_in,
                              void* d_out, int out_size, void* d_ws, size_t ws_size,
                              hipStream_t stream) {
    const float* emb  = (const float*)d_in[0];
    const int*   eidx = (const int*)d_in[1];
    const int*   batch= (const int*)d_in[2];
    const float* W1   = (const float*)d_in[3];
    const float* b1   = (const float*)d_in[4];
    const float* W2   = (const float*)d_in[5];
    const float* b2   = (const float*)d_in[6];
    const float* Wfc  = (const float*)d_in[7];
    const float* bfc  = (const float*)d_in[8];
    float* out = (float*)d_out;

    const int* src = eidx;             // edge_index[0]
    const int* dst = eidx + N_EDGES;   // edge_index[1]

    // ---- workspace partition (256B aligned slabs) ----
    char* w = (char*)d_ws;
    size_t off = 0;
    auto take = [&](size_t bytes) -> void* {
        void* p = w + off;
        off += (bytes + 255) & ~(size_t)255;
        return p;
    };
    unsigned short* emb_bf = (unsigned short*)take((size_t)N_NODES * IN_DIM * 2);  // 76.8 MB
    unsigned short* w1t    = (unsigned short*)take((size_t)HIDDEN * IN_DIM * 2);
    unsigned short* w2t    = (unsigned short*)take((size_t)HIDDEN * HIDDEN * 2);
    float*          h      = (float*)take((size_t)N_NODES * HIDDEN * 4);           // H1 then H2
    float*          g      = (float*)take((size_t)N_NODES * HIDDEN * 4);           // agg / x
    unsigned short* x_bf   = (unsigned short*)take((size_t)N_NODES * HIDDEN * 2);
    float*          deg    = (float*)take((size_t)N_NODES * 4);
    float*          dinv   = (float*)take((size_t)N_NODES * 4);
    float*          pooled = (float*)take((size_t)N_GRAPHS * HIDDEN * 4);
    float*          cnt    = (float*)take((size_t)N_GRAPHS * 4);
    (void)ws_size; (void)n_in; (void)in_sizes; (void)out_size;

    const long long nFeat = (long long)N_NODES * HIDDEN;

    // ---- prep: bf16 conversions + weight transposes ----
    cvt_f32_bf16<<<4096, 256, 0, stream>>>(emb, emb_bf, (long long)N_NODES * IN_DIM);
    w_transpose_bf16<<<(IN_DIM * HIDDEN + 255) / 256, 256, 0, stream>>>(W1, w1t, IN_DIM, HIDDEN);
    w_transpose_bf16<<<(HIDDEN * HIDDEN + 255) / 256, 256, 0, stream>>>(W2, w2t, HIDDEN, HIDDEN);

    // ---- degrees / symmetric norm ----
    deg_init<<<(N_NODES + 255) / 256, 256, 0, stream>>>(deg, N_NODES);
    deg_edges<<<(N_EDGES + 255) / 256, 256, 0, stream>>>(dst, deg, N_EDGES);
    dinv_kernel<<<(N_NODES + 255) / 256, 256, 0, stream>>>(deg, dinv, N_NODES);

    const int gemm1_blocks = (N_NODES / 16) * (HIDDEN / 16) / 8;  // 6250

    // ---- layer 1: H1 = emb @ W1 ; agg ; +b1 ; relu ----
    gemm_bf16_wmma<<<gemm1_blocks, 256, 0, stream>>>(emb_bf, w1t, h, N_NODES, HIDDEN, IN_DIM);
    zero_f32<<<4096, 256, 0, stream>>>(g, nFeat);
    agg_edges<<<(N_EDGES + 7) / 8, 256, 0, stream>>>(h, src, dst, dinv, g, N_EDGES);
    self_loops<<<N_NODES, HIDDEN, 0, stream>>>(h, dinv, g);
    bias_act<<<N_NODES, HIDDEN, 0, stream>>>(g, b1, 1);

    // ---- layer 2: H2 = x @ W2 ; agg ; +b2 ----
    cvt_f32_bf16<<<4096, 256, 0, stream>>>(g, x_bf, nFeat);
    gemm_bf16_wmma<<<gemm1_blocks, 256, 0, stream>>>(x_bf, w2t, h, N_NODES, HIDDEN, HIDDEN);
    zero_f32<<<4096, 256, 0, stream>>>(g, nFeat);
    agg_edges<<<(N_EDGES + 7) / 8, 256, 0, stream>>>(h, src, dst, dinv, g, N_EDGES);
    self_loops<<<N_NODES, HIDDEN, 0, stream>>>(h, dinv, g);
    bias_act<<<N_NODES, HIDDEN, 0, stream>>>(g, b2, 0);

    // ---- mean pool + FC head ----
    zero_f32<<<((N_GRAPHS * HIDDEN) + 255) / 256, 256, 0, stream>>>(pooled, (long long)N_GRAPHS * HIDDEN);
    zero_f32<<<1, 256, 0, stream>>>(cnt, N_GRAPHS);
    pool_nodes<<<N_NODES, HIDDEN, 0, stream>>>(g, batch, pooled, cnt);
    final_fc<<<N_GRAPHS, 32, 0, stream>>>(pooled, cnt, Wfc, bfc, out);
}
